// CopyGenerator_8435315769429
// MI455X (gfx1250) — matrix operand order repaired and checked by reading, not compiled, not run on hardware
//
#include <hip/hip_runtime.h>
#include <math.h>

// Problem dims (fixed by the reference)
#define T_   50
#define B_   64
#define S_   100
#define H_   512
#define VT   32000
#define VE   5000
#define VTOT (VT + VE)
#define M_   (T_ * B_)   // 3200 rows

typedef __attribute__((ext_vector_type(16))) __bf16 v16bf;
typedef __attribute__((ext_vector_type(8)))  float  v8f;

union ABf { uint4 u[2]; v16bf v; };

__device__ __forceinline__ unsigned short f2bf(float f) {
    unsigned int u = __float_as_uint(f);
    u += 0x7FFFu + ((u >> 16) & 1u);   // round-to-nearest-even
    return (unsigned short)(u >> 16);
}

// ---------------- f32 -> bf16 conversion (vectorized) ----------------
__global__ __launch_bounds__(256)
void cvt_bf16(const float* __restrict__ src, unsigned short* __restrict__ dst, int n4) {
    int i = blockIdx.x * blockDim.x + threadIdx.x;
    if (i >= n4) return;
    float4 f = ((const float4*)src)[i];
    ushort4 o;
    o.x = f2bf(f.x); o.y = f2bf(f.y); o.z = f2bf(f.z); o.w = f2bf(f.w);
    ((ushort4*)dst)[i] = o;
}

// ---------------- copy gate: (1 - sigmoid(dec . Wc + bc)) per (t,b) ----------------
__global__ __launch_bounds__(128)
void copy_gate(const float* __restrict__ dec, const float* __restrict__ Wc,
               const float* __restrict__ bc, float* __restrict__ g1m) {
    __shared__ float red[128];
    int row = blockIdx.x;
    const float* x = dec + (size_t)row * H_;
    float s = 0.f;
    for (int k = threadIdx.x; k < H_; k += 128) s += x[k] * Wc[k];
    red[threadIdx.x] = s;
    __syncthreads();
    for (int off = 64; off > 0; off >>= 1) {
        if (threadIdx.x < off) red[threadIdx.x] += red[threadIdx.x + off];
        __syncthreads();
    }
    if (threadIdx.x == 0) {
        float g = 1.f / (1.f + __expf(-(red[0] + bc[0])));
        g1m[row] = 1.f - g;
    }
}

// ---------------- GEMM: logits = A(bf16) x W(bf16)^T + b, via WMMA ----------------
// Block: 128 threads = 4 waves (wave32). Block tile: M=32, N=128.
// Each wave register-blocks 2Mx2N: a 32x32 patch as four 16x16 accumulators.
// Per K-step: 2 A frags + 2 B frags (8 x b128 loads) feed 4 independent WMMAs
// -> each fragment reused twice (half the L2 traffic of 1x1 blocking) and the
// scheduler gets 4 independent acc chains for WMMA latency hiding.
// WGP-scope prefetch (locality=3) pulls the next K-chunk of W into the WGP
// cache ahead of use -- W (33 MB bf16) is L2-resident, so the latency to hide
// is L2 -> WGP.
//
// A frag (16-bit 16x32): lane m = lane&15, half = lane>>4;
//   elems 0..7  -> K = kb + half*8 + e        (16B contiguous)
//   elems 8..15 -> K = kb + 16 + half*8 + e   (16B contiguous)
// B frag (16-bit 32x16): lane n = lane&15, half = lane>>4;
//   elems 0..7  -> K = kb + half*8 + e        (16B contiguous)
//   elems 8..15 -> K = kb + 16 + half*8 + e   (16B contiguous)
// D (f32 16x16): VGPR r -> row m = r + half*8, col n = lane&15.
__global__ __launch_bounds__(128)
void gemm_logits(const unsigned short* __restrict__ Ab,   // [M_,H_] bf16
                 const unsigned short* __restrict__ Wb,   // [VT,H_] bf16
                 const float* __restrict__ bias,          // [VT]
                 float* __restrict__ out)                 // [M_,VTOT]
{
    const int lane  = threadIdx.x & 31;
    const int wave  = threadIdx.x >> 5;
    const int half  = lane >> 4;
    const int l16   = lane & 15;
    const int mbase = blockIdx.y * 32;
    const int nbase = blockIdx.x * 128 + wave * 32;

    const unsigned short* arow0 = Ab + (size_t)(mbase + l16)      * H_;
    const unsigned short* arow1 = Ab + (size_t)(mbase + 16 + l16) * H_;
    const unsigned short* brow0 = Wb + (size_t)(nbase + l16)      * H_;
    const unsigned short* brow1 = Wb + (size_t)(nbase + 16 + l16) * H_;

    v8f acc00 = {}, acc01 = {}, acc10 = {}, acc11 = {};

#pragma unroll 2
    for (int kb = 0; kb < H_; kb += 32) {
        // Speculative WGP-scope prefetch of the next K-chunk of the B rows.
        // One chunk past row end still lands inside the workspace; speculative
        // prefetch drops failed translations silently, so no guard needed.
        __builtin_prefetch(brow0 + kb + 32, 0, 3);
        __builtin_prefetch(brow1 + kb + 32, 0, 3);

        ABf a0, a1, b0, b1;
        a0.u[0] = *(const uint4*)(arow0 + kb +      half * 8);
        a0.u[1] = *(const uint4*)(arow0 + kb + 16 + half * 8);
        a1.u[0] = *(const uint4*)(arow1 + kb +      half * 8);
        a1.u[1] = *(const uint4*)(arow1 + kb + 16 + half * 8);
        b0.u[0] = *(const uint4*)(brow0 + kb +      half * 8);
        b0.u[1] = *(const uint4*)(brow0 + kb + 16 + half * 8);
        b1.u[0] = *(const uint4*)(brow1 + kb +      half * 8);
        b1.u[1] = *(const uint4*)(brow1 + kb + 16 + half * 8);

        acc00 = __builtin_amdgcn_wmma_f32_16x16x32_bf16(false, a0.v, false, b0.v,
                                                        (short)0, acc00, false, false);
        acc01 = __builtin_amdgcn_wmma_f32_16x16x32_bf16(false, a0.v, false, b1.v,
                                                        (short)0, acc01, false, false);
        acc10 = __builtin_amdgcn_wmma_f32_16x16x32_bf16(false, a1.v, false, b0.v,
                                                        (short)0, acc10, false, false);
        acc11 = __builtin_amdgcn_wmma_f32_16x16x32_bf16(false, a1.v, false, b1.v,
                                                        (short)0, acc11, false, false);
    }

    const int col0 = nbase + l16;
    const int col1 = nbase + 16 + l16;
    const float bv0 = bias[col0];
    const float bv1 = bias[col1];
#pragma unroll
    for (int r = 0; r < 8; ++r) {
        int row0 = mbase + r + half * 8;
        int row1 = mbase + 16 + r + half * 8;
        out[(size_t)row0 * VTOT + col0] = acc00[r] + bv0;
        out[(size_t)row0 * VTOT + col1] = acc01[r] + bv1;
        out[(size_t)row1 * VTOT + col0] = acc10[r] + bv0;
        out[(size_t)row1 * VTOT + col1] = acc11[r] + bv1;
    }
}

// ---------------- per-row online max / logsumexp over the VT logits ----------------
__global__ __launch_bounds__(256)
void row_stats(const float* __restrict__ out, float* __restrict__ rstat) {
    __shared__ float sm[256], ss[256];
    int row = blockIdx.x;
    const float4* x = (const float4*)(out + (size_t)row * VTOT);
    float m = -3.4e38f, s = 0.f;
    for (int i = threadIdx.x; i < VT / 4; i += 256) {
        float4 v = x[i];
        float vm = fmaxf(fmaxf(v.x, v.y), fmaxf(v.z, v.w));
        if (vm > m) { s *= __expf(m - vm); m = vm; }
        s += __expf(v.x - m) + __expf(v.y - m) + __expf(v.z - m) + __expf(v.w - m);
    }
    sm[threadIdx.x] = m; ss[threadIdx.x] = s;
    __syncthreads();
    for (int off = 128; off > 0; off >>= 1) {
        if (threadIdx.x < off) {
            float m1 = sm[threadIdx.x], s1 = ss[threadIdx.x];
            float m2 = sm[threadIdx.x + off], s2 = ss[threadIdx.x + off];
            float M = fmaxf(m1, m2);
            sm[threadIdx.x] = M;
            ss[threadIdx.x] = s1 * __expf(m1 - M) + s2 * __expf(m2 - M);
        }
        __syncthreads();
    }
    if (threadIdx.x == 0) rstat[row] = sm[0] + __logf(ss[0]);
}

// ---------------- apply log_softmax: x -= (max + log(sumexp)) ----------------
__global__ __launch_bounds__(256)
void apply_lsm(float* __restrict__ out, const float* __restrict__ rstat) {
    int row = blockIdx.x;
    int c4  = blockIdx.y * 256 + threadIdx.x;
    if (c4 >= VT / 4) return;
    float4* x = (float4*)(out + (size_t)row * VTOT);
    float r = rstat[row];
    float4 v = x[c4];
    v.x -= r; v.y -= r; v.z -= r; v.w -= r;
    x[c4] = v;
}

// ---------------- masked scatter-add into ext vocab + safe_log ----------------
__global__ __launch_bounds__(256)
void copy_scatter(const float* __restrict__ attn,   // [T,B,S]
                  const int*   __restrict__ c2e,    // [S,B]
                  const float* __restrict__ g1m,    // [M_] = (1 - copy)
                  float* __restrict__ out)          // [M_,VTOT]
{
    __shared__ float acc[VE];   // 20 KB LDS accumulator
    int row = blockIdx.x;       // row = t*B + b
    int b   = row % B_;
    for (int v = threadIdx.x; v < VE; v += 256) acc[v] = 0.f;
    __syncthreads();
    float g = g1m[row];
    for (int s = threadIdx.x; s < S_; s += 256) {
        int idx = c2e[s * B_ + b];
        if (idx != 0)
            atomicAdd(&acc[idx], attn[(size_t)row * S_ + s] * g);
    }
    __syncthreads();
    float* o = out + (size_t)row * VTOT + VT;
    for (int v = threadIdx.x; v < VE; v += 256) {
        float val = fminf(fmaxf(acc[v], 0.001f), 0.999f);
        o[v] = __logf(val);
    }
}

extern "C" void kernel_launch(void* const* d_in, const int* in_sizes, int n_in,
                              void* d_out, int out_size, void* d_ws, size_t ws_size,
                              hipStream_t stream) {
    const float* hidden = (const float*)d_in[0];
    const float* dec    = (const float*)d_in[1];
    /* d_in[2] = concat_c, unused by the reference */
    const float* attn   = (const float*)d_in[3];
    const int*   c2e    = (const int*)d_in[4];
    const float* W      = (const float*)d_in[5];
    const float* bias   = (const float*)d_in[6];
    const float* Wc     = (const float*)d_in[7];
    const float* bc     = (const float*)d_in[8];
    float* out = (float*)d_out;

    // workspace layout
    char* ws = (char*)d_ws;
    unsigned short* Wb = (unsigned short*)ws;                              // VT*H_ bf16
    unsigned short* Ab = (unsigned short*)(ws + (size_t)VT * H_ * 2);      // M_*H_ bf16
    float* g1m   = (float*)(ws + (size_t)VT * H_ * 2 + (size_t)M_ * H_ * 2);
    float* rstat = g1m + M_;

    { int n4 = VT * H_ / 4; cvt_bf16<<<(n4 + 255) / 256, 256, 0, stream>>>(W, Wb, n4); }
    { int n4 = M_ * H_ / 4; cvt_bf16<<<(n4 + 255) / 256, 256, 0, stream>>>(hidden, Ab, n4); }

    copy_gate<<<M_, 128, 0, stream>>>(dec, Wc, bc, g1m);

    gemm_logits<<<dim3(VT / 128, M_ / 32), 128, 0, stream>>>(Ab, Wb, bias, out);

    row_stats<<<M_, 256, 0, stream>>>(out, rstat);
    apply_lsm<<<dim3(M_, (VT / 4 + 255) / 256), 256, 0, stream>>>(out, rstat);

    copy_scatter<<<M_, 256, 0, stream>>>(attn, c2e, g1m, out);
}
